// SelfAttention2D_61735859912981
// MI455X (gfx1250) — matrix operand order repaired and checked by reading, not compiled, not run on hardware
//
#include <hip/hip_runtime.h>

typedef _Float16 v16h __attribute__((ext_vector_type(16)));
typedef _Float16 v8h  __attribute__((ext_vector_type(8)));
typedef float    v8f  __attribute__((ext_vector_type(8)));
typedef int      v4i  __attribute__((ext_vector_type(4)));

#define C_DIM  512
#define HW     1024
#define BATCH  8
#define HEADS  8
#define HDIM   64
#define GROUPS 32
#define CPG    16
#define EPS    1e-5f

#if __has_builtin(__builtin_amdgcn_global_load_async_to_lds_b128)
#define HAVE_ASYNC_LDS 1
#endif

__device__ __forceinline__ void copy16(const _Float16* g, _Float16* l) {
#ifdef HAVE_ASYNC_LDS
    __builtin_amdgcn_global_load_async_to_lds_b128(
        (__attribute__((address_space(1))) v4i*)g,
        (__attribute__((address_space(3))) v4i*)l, 0, 0);
#else
    *(v8h*)l = *(const v8h*)g;
#endif
}

__device__ __forceinline__ void wait_async() {
#ifdef HAVE_ASYNC_LDS
#if __has_builtin(__builtin_amdgcn_s_wait_asynccnt)
    __builtin_amdgcn_s_wait_asynccnt(0);
#else
    asm volatile("s_wait_asynccnt 0x0" ::: "memory");
#endif
#endif
}

__device__ __forceinline__ v16h make_v16(v8h lo, v8h hi) {
    union { v16h v; v8h h[2]; } u;
    u.h[0] = lo; u.h[1] = hi;
    return u.v;
}

// A-fragment (16xK slice, K chunk of 32): row = rowBase + (lane&15),
// 16 contiguous K values starting at kBase + 16*(lane>=16)
__device__ __forceinline__ v16h load_afrag(const _Float16* A, int rowBase, int ld,
                                           int kBase, int lo16, int hi) {
    const _Float16* p = A + (size_t)(rowBase + lo16) * ld + kBase + hi * 16;
    v8h a0 = *(const v8h*)(p);
    v8h a1 = *(const v8h*)(p + 8);
    return make_v16(a0, a1);
}

// B-fragment (Kx16 slice): column = colBase + (lane&15); 16 contiguous K
// values starting at kBase + 16*(lane>=16). Bm[col][k], k contiguous (ld).
__device__ __forceinline__ v16h load_bfrag(const _Float16* Bm, int colBase, int ld,
                                           int kBase, int lo16, int hi) {
    const _Float16* p = Bm + (size_t)(colBase + lo16) * ld + kBase + hi * 16;
    return *(const v16h*)(p);
}

#define WMMA_F16(a, b, c) __builtin_amdgcn_wmma_f32_16x16x32_f16(false, a, false, b, (short)0, c, false, false)

// ---------------------------------------------------------------- kernel 0
__global__ void k_wconvert(const float* __restrict__ qkv_w,
                           const float* __restrict__ proj_w,
                           _Float16* __restrict__ wq, _Float16* __restrict__ wp) {
    int i = blockIdx.x * blockDim.x + threadIdx.x;
    if (i < 3 * C_DIM * C_DIM) wq[i] = (_Float16)qkv_w[i];
    if (i < C_DIM * C_DIM)     wp[i] = (_Float16)proj_w[i];
}

// ---------------------------------------------------------------- kernel 1
// GroupNorm; writes normalized activations transposed as (B, HW, C) f16.
__global__ void k_gnorm(const float* __restrict__ x, const float* __restrict__ w,
                        const float* __restrict__ bgn, _Float16* __restrict__ Xnt) {
    __shared__ float s1[256], s2[256];
    const int t = threadIdx.x;
    const int b = blockIdx.x >> 5;
    const int g = blockIdx.x & 31;
    const float* xb = x + ((size_t)b * C_DIM + g * CPG) * HW;
    float a = 0.f, q = 0.f;
    for (int i = t; i < CPG * HW; i += 256) {
        float v = xb[i];
        a += v; q += v * v;
    }
    s1[t] = a; s2[t] = q;
    __syncthreads();
    for (int st = 128; st > 0; st >>= 1) {
        if (t < st) { s1[t] += s1[t + st]; s2[t] += s2[t + st]; }
        __syncthreads();
    }
    const float mu  = s1[0] * (1.0f / (CPG * HW));
    const float var = s2[0] * (1.0f / (CPG * HW)) - mu * mu;
    const float inv = rsqrtf(var + EPS);
    for (int i = t; i < CPG * HW; i += 256) {
        int c = i >> 10, n = i & 1023;
        int ch = g * CPG + c;
        float v = (xb[i] - mu) * inv * w[ch] + bgn[ch];
        Xnt[((size_t)b * HW + n) * C_DIM + ch] = (_Float16)v;
    }
}

// -------------------------------------------------- shared GEMM core (32x64)
// Software-pipelined: prefetch next K-chunk fragments while current 8 WMMAs
// issue, so the scheduler can use partial s_wait_loadcnt instead of wait-0.
__device__ __forceinline__ void gemm_core_32x64(const _Float16* __restrict__ W, int o0,
                                                const _Float16* __restrict__ Bm, int n0,
                                                int lo16, int hi, v8f acc[2][4]) {
    v16h a0 = load_afrag(W, o0,      C_DIM, 0, lo16, hi);
    v16h a1 = load_afrag(W, o0 + 16, C_DIM, 0, lo16, hi);
    v16h bb[4];
#pragma unroll
    for (int jn = 0; jn < 4; ++jn)
        bb[jn] = load_bfrag(Bm, n0 + 16 * jn, C_DIM, 0, lo16, hi);
#pragma unroll
    for (int kc = 0; kc < C_DIM; kc += 32) {
        v16h na0 = a0, na1 = a1;
        v16h nb[4] = { bb[0], bb[1], bb[2], bb[3] };
        if (kc + 32 < C_DIM) {
            na0 = load_afrag(W, o0,      C_DIM, kc + 32, lo16, hi);
            na1 = load_afrag(W, o0 + 16, C_DIM, kc + 32, lo16, hi);
#pragma unroll
            for (int jn = 0; jn < 4; ++jn)
                nb[jn] = load_bfrag(Bm, n0 + 16 * jn, C_DIM, kc + 32, lo16, hi);
        }
#pragma unroll
        for (int jn = 0; jn < 4; ++jn) {
            acc[0][jn] = WMMA_F16(a0, bb[jn], acc[0][jn]);
            acc[1][jn] = WMMA_F16(a1, bb[jn], acc[1][jn]);
        }
        a0 = na0; a1 = na1;
#pragma unroll
        for (int jn = 0; jn < 4; ++jn) bb[jn] = nb[jn];
    }
}

// ---------------------------------------------------------------- kernel 2
// QKV GEMM: 32(M=o) x 64(N=n) tile per wave.
// Epilogue scatters into Q/K as (b,h,n,d) and V as (b,h,d,m), f16.
__global__ void k_gemm_qkv(const _Float16* __restrict__ W,
                           const _Float16* __restrict__ Xnt,
                           const float* __restrict__ bias,
                           _Float16* __restrict__ Qo, _Float16* __restrict__ Ko,
                           _Float16* __restrict__ Vo) {
    const int wave = threadIdx.x >> 5;
    const int lane = threadIdx.x & 31;
    const int lo16 = lane & 15, hi = lane >> 4;
    int gid = blockIdx.x * 8 + wave;
    int b   = gid / (48 * 16);
    int rem = gid % (48 * 16);
    int mt = rem / 16, nt = rem % 16;
    int o0 = mt * 32, n0 = nt * 64;
    const _Float16* Xb = Xnt + (size_t)b * HW * C_DIM;
    v8f acc[2][4] = {};
    gemm_core_32x64(W, o0, Xb, n0, lo16, hi, acc);
#pragma unroll
    for (int im = 0; im < 2; ++im) {
        const int os   = o0 + 16 * im;
        const int sect = os >> 9;          // 0=q,1=k,2=v
        const int oin  = os & 511;
        const int head = oin >> 6;
        const int d0   = oin & 63;
        const size_t bh = (size_t)b * HEADS + head;
#pragma unroll
        for (int jn = 0; jn < 4; ++jn) {
            const int n = n0 + 16 * jn + lo16;
            if (sect < 2) {
                union { v8h v; _Float16 e[8]; } u;
#pragma unroll
                for (int r = 0; r < 8; ++r)
                    u.e[r] = (_Float16)(acc[im][jn][r] + bias[os + r + 8 * hi]);
                _Float16* dst = (sect == 0) ? Qo : Ko;
                *(v8h*)(dst + (bh * HW + n) * HDIM + d0 + 8 * hi) = u.v;
            } else {
#pragma unroll
                for (int r = 0; r < 8; ++r) {
                    int d = d0 + r + 8 * hi;
                    Vo[(bh * HDIM + d) * HW + n] =
                        (_Float16)(acc[im][jn][r] + bias[os + r + 8 * hi]);
                }
            }
        }
    }
}

// ---------------------------------------------------------------- kernel 3
// Flash attention. 4 waves/WG share (b,head); K/V chunks double-buffered in
// LDS via async load-to-LDS so the DMA for step j+1 overlaps WMMAs of step j.
#define NW 4
#define KLD 72   // 64 + 8 pad (144B row: 36-bank stride, cycle 16)
#define VLD 40   // 32 + 8 pad (80B row: 20-bank stride, cycle 16)

__device__ __forceinline__ void stage_kv(const _Float16* Kt, const _Float16* Vb, int m0,
                                         _Float16 (*kbuf)[KLD], _Float16 (*vbuf)[VLD],
                                         int krow0, int kseg0, int krow1, int kseg1,
                                         int vrow, int vseg) {
    copy16(Kt + (size_t)(m0 + krow0) * HDIM + kseg0,     &kbuf[krow0][kseg0]);
    copy16(Kt + (size_t)(m0 + krow0) * HDIM + kseg0 + 8, &kbuf[krow0][kseg0 + 8]);
    copy16(Kt + (size_t)(m0 + krow1) * HDIM + kseg1,     &kbuf[krow1][kseg1]);
    copy16(Kt + (size_t)(m0 + krow1) * HDIM + kseg1 + 8, &kbuf[krow1][kseg1 + 8]);
    const _Float16* vg = Vb + (size_t)vrow * HW + m0 + vseg;
    copy16(vg,     &vbuf[vrow][vseg]);
    copy16(vg + 8, &vbuf[vrow][vseg + 8]);
}

__global__ void k_attn(const _Float16* __restrict__ Q, const _Float16* __restrict__ Kb,
                       const _Float16* __restrict__ V, _Float16* __restrict__ AOt) {
    __shared__ __align__(64) _Float16 plds[NW][16][VLD];
    __shared__ __align__(64) _Float16 kls[2][32][KLD];
    __shared__ __align__(64) _Float16 vls[2][64][VLD];
    const int t    = threadIdx.x;
    const int wave = t >> 5;
    const int lane = t & 31;
    const int lo16 = lane & 15, hi = lane >> 4;
    const int bh = blockIdx.x;                 // b*8+head
    const int nt = blockIdx.y * NW + wave;     // 0..63
    const int b = bh >> 3, head = bh & 7;
    const int n0 = nt * 16;
    const _Float16* Qb = Q  + (size_t)bh * HW * HDIM;
    const _Float16* Kt = Kb + (size_t)bh * HW * HDIM;
    const _Float16* Vb = V  + (size_t)bh * HDIM * HW;

    // Q A-fragments (K-dim = d, 2 chunks of 32), loaded once
    v16h qa0 = load_afrag(Qb, n0, HDIM, 0,  lo16, hi);
    v16h qa1 = load_afrag(Qb, n0, HDIM, 32, lo16, hi);

    v8f ov0 = {}, ov1 = {}, ov2 = {}, ov3 = {};
    float mrow[8], lrow[8];
#pragma unroll
    for (int r = 0; r < 8; ++r) { mrow[r] = -1e30f; lrow[r] = 0.f; }
    const float scale = 0.125f;    // 1/sqrt(64)

    // staging maps (fixed per thread): K rows 0..31 x 64; V rows 0..63 x 32
    const int krow0 = t >> 2,  kseg0 = (t & 3) * 16;
    const int krow1 = (t + 128) >> 2, kseg1 = ((t + 128) & 3) * 16;
    const int vrow = t >> 1, vseg = (t & 1) * 16;

    // prologue: stage chunk 0 into buffer 0
    stage_kv(Kt, Vb, 0, kls[0], vls[0], krow0, kseg0, krow1, kseg1, vrow, vseg);
    wait_async();
    __syncthreads();

    for (int j = 0; j < HW / 32; ++j) {
        const int cur = j & 1;
        if (j + 1 < HW / 32)    // overlap next chunk's DMA with this chunk's math
            stage_kv(Kt, Vb, (j + 1) * 32, kls[cur ^ 1], vls[cur ^ 1],
                     krow0, kseg0, krow1, kseg1, vrow, vseg);

        v8f s0 = {}, s1 = {};
        {
            const _Float16* kbase = &kls[cur][0][0];
            v16h kb;
            kb = load_bfrag(kbase, 0,  KLD, 0,  lo16, hi);
            s0 = WMMA_F16(qa0, kb, s0);
            kb = load_bfrag(kbase, 0,  KLD, 32, lo16, hi);
            s0 = WMMA_F16(qa1, kb, s0);
            kb = load_bfrag(kbase, 16, KLD, 0,  lo16, hi);
            s1 = WMMA_F16(qa0, kb, s1);
            kb = load_bfrag(kbase, 16, KLD, 32, lo16, hi);
            s1 = WMMA_F16(qa1, kb, s1);
        }
        float p0[8], p1[8], alpha[8];
#pragma unroll
        for (int r = 0; r < 8; ++r) {
            float a = s0[r] * scale, c = s1[r] * scale;
            float mx = fmaxf(a, c);
#pragma unroll
            for (int msk = 1; msk < 16; msk <<= 1)
                mx = fmaxf(mx, __shfl_xor(mx, msk, 32));
            float mnew = fmaxf(mrow[r], mx);
            float e0 = __expf(a - mnew), e1 = __expf(c - mnew);
            float rs = e0 + e1;
#pragma unroll
            for (int msk = 1; msk < 16; msk <<= 1)
                rs += __shfl_xor(rs, msk, 32);
            float al = __expf(mrow[r] - mnew);
            lrow[r] = lrow[r] * al + rs;
            mrow[r] = mnew;
            alpha[r] = al;
            p0[r] = e0; p1[r] = e1;
        }
#pragma unroll
        for (int r = 0; r < 8; ++r) {
            ov0[r] *= alpha[r]; ov1[r] *= alpha[r];
            ov2[r] *= alpha[r]; ov3[r] *= alpha[r];
        }
        // transpose P through per-wave LDS tile -> A-fragment layout
#pragma unroll
        for (int r = 0; r < 8; ++r) {
            plds[wave][r + 8 * hi][lo16]      = (_Float16)p0[r];
            plds[wave][r + 8 * hi][16 + lo16] = (_Float16)p1[r];
        }
        __syncthreads();
        v16h pa = load_afrag(&plds[wave][0][0], 0, VLD, 0, lo16, hi);
        {
            const _Float16* vbase = &vls[cur][0][0];
            v16h vb;
            vb  = load_bfrag(vbase, 0,  VLD, 0, lo16, hi);
            ov0 = WMMA_F16(pa, vb, ov0);
            vb  = load_bfrag(vbase, 16, VLD, 0, lo16, hi);
            ov1 = WMMA_F16(pa, vb, ov1);
            vb  = load_bfrag(vbase, 32, VLD, 0, lo16, hi);
            ov2 = WMMA_F16(pa, vb, ov2);
            vb  = load_bfrag(vbase, 48, VLD, 0, lo16, hi);
            ov3 = WMMA_F16(pa, vb, ov3);
        }
        wait_async();       // my async copies for j+1 have landed
        __syncthreads();    // everyone's copies landed; my reads of buf[cur] done
    }
    // normalize and store transposed: AOt[b][n][head*64 + d] f16
    _Float16* dst = AOt + (size_t)b * HW * C_DIM + (size_t)head * HDIM;
#pragma unroll
    for (int r = 0; r < 8; ++r) {
        float inv = 1.0f / lrow[r];
        int n = n0 + r + 8 * hi;
        _Float16* row = dst + (size_t)n * C_DIM;
        row[0  + lo16] = (_Float16)(ov0[r] * inv);
        row[16 + lo16] = (_Float16)(ov1[r] * inv);
        row[32 + lo16] = (_Float16)(ov2[r] * inv);
        row[48 + lo16] = (_Float16)(ov3[r] * inv);
    }
}

// ---------------------------------------------------------------- kernel 4
// Output projection + bias + residual, f32 output. 32x64 tile per wave.
__global__ void k_gemm_proj(const _Float16* __restrict__ W,
                            const _Float16* __restrict__ Bt,
                            const float* __restrict__ bias,
                            const float* __restrict__ x,
                            float* __restrict__ out) {
    const int wave = threadIdx.x >> 5;
    const int lane = threadIdx.x & 31;
    const int lo16 = lane & 15, hi = lane >> 4;
    int gid = blockIdx.x * 8 + wave;
    int b   = gid / (16 * 16);
    int rem = gid % (16 * 16);
    int mt = rem / 16, nt = rem % 16;
    int o0 = mt * 32, n0 = nt * 64;
    const _Float16* Bb = Bt + (size_t)b * HW * C_DIM;
    v8f acc[2][4] = {};
    gemm_core_32x64(W, o0, Bb, n0, lo16, hi, acc);
#pragma unroll
    for (int im = 0; im < 2; ++im) {
#pragma unroll
        for (int jn = 0; jn < 4; ++jn) {
            const int n = n0 + 16 * jn + lo16;
#pragma unroll
            for (int r = 0; r < 8; ++r) {
                int o = o0 + 16 * im + r + 8 * hi;
                size_t idx = ((size_t)b * C_DIM + o) * HW + n;
                out[idx] = x[idx] + acc[im][jn][r] + bias[o];
            }
        }
    }
}

// ---------------------------------------------------------------- launch
extern "C" void kernel_launch(void* const* d_in, const int* in_sizes, int n_in,
                              void* d_out, int out_size, void* d_ws, size_t ws_size,
                              hipStream_t stream) {
    const float* x      = (const float*)d_in[0];
    const float* gn_w   = (const float*)d_in[1];
    const float* gn_b   = (const float*)d_in[2];
    const float* qkv_w  = (const float*)d_in[3];
    const float* qkv_b  = (const float*)d_in[4];
    const float* proj_w = (const float*)d_in[5];
    const float* proj_b = (const float*)d_in[6];
    float* out = (float*)d_out;

    char* ws = (char*)d_ws;
    size_t off = 0;
    auto alloc = [&](size_t bytes) {
        off = (off + 255) & ~(size_t)255;
        char* p = ws + off;
        off += bytes;
        return p;
    };
    _Float16* Wq  = (_Float16*)alloc((size_t)3 * C_DIM * C_DIM * 2);
    _Float16* Wp  = (_Float16*)alloc((size_t)C_DIM * C_DIM * 2);
    _Float16* Xnt = (_Float16*)alloc((size_t)BATCH * HW * C_DIM * 2);
    _Float16* Q   = (_Float16*)alloc((size_t)BATCH * HEADS * HW * HDIM * 2);
    _Float16* K   = (_Float16*)alloc((size_t)BATCH * HEADS * HW * HDIM * 2);
    _Float16* V   = (_Float16*)alloc((size_t)BATCH * HEADS * HW * HDIM * 2);
    _Float16* AOt = (_Float16*)alloc((size_t)BATCH * HW * C_DIM * 2);

    k_wconvert<<<3072, 256, 0, stream>>>(qkv_w, proj_w, Wq, Wp);
    k_gnorm<<<BATCH * GROUPS, 256, 0, stream>>>(x, gn_w, gn_b, Xnt);
    k_gemm_qkv<<<768, 256, 0, stream>>>(Wq, Xnt, qkv_b, Q, K, V);
    dim3 ag(BATCH * HEADS, 16);
    k_attn<<<ag, NW * 32, 0, stream>>>(Q, K, V, AOt);
    k_gemm_proj<<<256, 256, 0, stream>>>(Wp, AOt, proj_b, x, out);
}
